// Vanilla_LSTM_9500467659082
// MI455X (gfx1250) — compile-verified, matrix operand order checked by reading, not run on hardware
//
#include <hip/hip_runtime.h>
#include <hip/hip_bf16.h>

// ---------------------------------------------------------------------------
// LSTM LM forward on gfx1250 (CDNA5, wave32, WMMA bf16 -> f32 accumulate)
//   B=8, S=256, H=512, E=512, V=50257
// Data movement: GLOBAL_LOAD_ASYNC_TO_LDS_B128 staging (ASYNCcnt protocol).
// ---------------------------------------------------------------------------

#define Bc   8
#define Sc   256
#define Hc   512
#define Ec   512
#define Vc   50257
#define Gc   2048          // 4*H
#define Mrows 2048         // B*S, row order m = s*B + b
#define NPAD 50304         // 393 * 128  (V padded to BN multiple)

typedef __attribute__((ext_vector_type(16))) __bf16 v16bf;
typedef __attribute__((ext_vector_type(8)))  float  v8f;

union FragU { uint4 q[2]; v16bf v; };

__device__ __forceinline__ unsigned short f2bf(float x) {
    unsigned int u = __float_as_uint(x);
    unsigned int r = u + 0x7FFFu + ((u >> 16) & 1u);   // round-to-nearest-even
    return (unsigned short)(r >> 16);
}

__device__ __forceinline__ float sigm(float x) {
    return 1.0f / (1.0f + __expf(-x));
}

// Async copy 16B global -> LDS, one instruction per lane (GV mode, ASYNCcnt).
__device__ __forceinline__ void async_copy_b128(unsigned int lds_addr,
                                                const void* gaddr) {
    asm volatile("global_load_async_to_lds_b128 %0, %1, off"
                 :
                 : "v"(lds_addr), "v"((unsigned long long)(uintptr_t)gaddr)
                 : "memory");
}

__device__ __forceinline__ void wait_async_all() {
    asm volatile("s_wait_asynccnt 0x0" ::: "memory");
}

// Load one 16x32 bf16 WMMA fragment (A layout; B uses identical per-lane math
// on a pre-transposed [N][K] matrix). Per lane: two contiguous 16B chunks:
//   elements K = k0 + hi*8 .. +7   and   K = k0 + 16 + hi*8 .. +7
__device__ __forceinline__ v16bf load_frag(const unsigned short* base,
                                           int row, int stride, int k0, int hi) {
    FragU f;
    const unsigned short* p = base + (size_t)row * stride + k0 + hi * 8;
    f.q[0] = *(const uint4*)(p);
    f.q[1] = *(const uint4*)(p + 16);
    return f.v;
}

// ---------------------------------------------------------------------------
// Setup kernels: transpose + bf16-convert weights, gather embeddings, init.
// ---------------------------------------------------------------------------

__global__ void build_WUT(const float* __restrict__ Wf, const float* __restrict__ Wi,
                          const float* __restrict__ Wo, const float* __restrict__ Wc,
                          const float* __restrict__ Uf, const float* __restrict__ Ui,
                          const float* __restrict__ Uo, const float* __restrict__ Uc,
                          unsigned short* __restrict__ WxT, unsigned short* __restrict__ UhT) {
    int g = blockIdx.x;                 // 0..2047 (gate-major output column)
    int gate = g >> 9, u = g & 511;
    const float* W = (gate == 0) ? Wf : (gate == 1) ? Wi : (gate == 2) ? Wo : Wc;
    const float* U = (gate == 0) ? Uf : (gate == 1) ? Ui : (gate == 2) ? Uo : Uc;
    for (int e = threadIdx.x; e < Ec; e += 256) {
        WxT[(size_t)g * Ec + e] = f2bf(W[(size_t)e * Hc + u]);
        UhT[(size_t)g * Hc + e] = f2bf(U[(size_t)e * Hc + u]);
    }
}

__global__ void build_WoutT(const float* __restrict__ Wout,
                            unsigned short* __restrict__ WoutT) {
    int n = blockIdx.x * 128 + threadIdx.x;      // 0..NPAD-1
    unsigned short* dst = WoutT + (size_t)n * Hc;
    if (n < Vc) {
        for (int k = 0; k < Hc; ++k) dst[k] = f2bf(Wout[(size_t)k * Vc + n]);
    } else {
        for (int k = 0; k < Hc; ++k) dst[k] = 0;
    }
}

__global__ void embed_gather(const int* __restrict__ x, const float* __restrict__ emb,
                             unsigned short* __restrict__ xe) {
    int m = blockIdx.x;                          // m = s*B + b
    int s = m >> 3, b = m & 7;
    int tok = x[b * Sc + s];
    const float* src = emb + (size_t)tok * Hc;
    unsigned short* dst = xe + (size_t)m * Hc;
    for (int e = threadIdx.x; e < Hc; e += 128) dst[e] = f2bf(src[e]);
}

__global__ void init_bias(const float* __restrict__ bWf, const float* __restrict__ bUf,
                          const float* __restrict__ bWi, const float* __restrict__ bUi,
                          const float* __restrict__ bWo, const float* __restrict__ bUo,
                          const float* __restrict__ bWc, const float* __restrict__ bUc,
                          float* __restrict__ bias) {
    int g = blockIdx.x * 256 + threadIdx.x;      // 0..2047
    int gate = g >> 9, u = g & 511;
    const float* bW = (gate == 0) ? bWf : (gate == 1) ? bWi : (gate == 2) ? bWo : bWc;
    const float* bU = (gate == 0) ? bUf : (gate == 1) ? bUi : (gate == 2) ? bUo : bUc;
    bias[g] = bW[u] + bU[u];
}

__global__ void init_state(const float* __restrict__ h0, const float* __restrict__ c0,
                           unsigned short* __restrict__ hb0, unsigned short* __restrict__ hb1,
                           float* __restrict__ hF, float* __restrict__ cF) {
    int idx = blockIdx.x * 256 + threadIdx.x;    // 0..8191 (16 rows x 512)
    int row = idx >> 9;
    float v = (row < Bc) ? h0[idx] : 0.0f;       // rows 8..15 = zero padding
    unsigned short hv = f2bf(v);
    hb0[idx] = hv;
    hb1[idx] = hv;
    if (idx < Bc * Hc) { hF[idx] = h0[idx]; cF[idx] = c0[idx]; }
}

// ---------------------------------------------------------------------------
// Generic WMMA GEMM: C[M x N] (f32) = A[M x 512] (bf16, row-major)
//                                   x BT[N x 512]^T (bf16, pre-transposed).
// Block tile 128x128, K-chunk 32, 8 waves (each 64x32 = 4x2 wmma tiles).
// Tiles staged via GLOBAL_LOAD_ASYNC_TO_LDS_B128 (ASYNCcnt), fragments read
// back with ds_load_b128, math on v_wmma_f32_16x16x32_bf16.
// mode 0: plain store C[m*N + n]          (input projection -> xg)
// mode 1: logits store with row remap (m = s*B+b  ->  out row b*S+s),
//         +bout, column guard n < V.
// ---------------------------------------------------------------------------

__global__ __launch_bounds__(256) void gemm_bf16_wmma(
        const unsigned short* __restrict__ A, const unsigned short* __restrict__ BT,
        float* __restrict__ C, int Ntotal, int mode, const float* __restrict__ bout) {
    __shared__ unsigned short As[128 * 32];
    __shared__ unsigned short Bs[128 * 32];

    const int tid  = threadIdx.x;
    const int lane = tid & 31;
    const int wave = tid >> 5;
    const int wm = wave >> 2, wn = wave & 3;     // 2 x 4 wave grid
    const int r  = lane & 15, hi = lane >> 4;
    const int m0 = blockIdx.y * 128;
    const int n0 = blockIdx.x * 128;

    // Per-thread staging slots: rows (tid>>2) and (tid>>2)+64, quarter tid&3.
    const int srow = tid >> 2, sq = tid & 3;
    const unsigned int ldsA0 = (unsigned int)(uintptr_t)&As[srow * 32 + sq * 8];
    const unsigned int ldsA1 = (unsigned int)(uintptr_t)&As[(srow + 64) * 32 + sq * 8];
    const unsigned int ldsB0 = (unsigned int)(uintptr_t)&Bs[srow * 32 + sq * 8];
    const unsigned int ldsB1 = (unsigned int)(uintptr_t)&Bs[(srow + 64) * 32 + sq * 8];
    const unsigned short* gA0 = A  + (size_t)(m0 + srow)      * 512 + sq * 8;
    const unsigned short* gA1 = A  + (size_t)(m0 + srow + 64) * 512 + sq * 8;
    const unsigned short* gB0 = BT + (size_t)(n0 + srow)      * 512 + sq * 8;
    const unsigned short* gB1 = BT + (size_t)(n0 + srow + 64) * 512 + sq * 8;

    v8f acc[4][2];
#pragma unroll
    for (int im = 0; im < 4; ++im)
#pragma unroll
        for (int jn = 0; jn < 2; ++jn) acc[im][jn] = (v8f){};

    for (int k0 = 0; k0 < 512; k0 += 32) {
        // Async-stage 128x32 tiles of A and BT into LDS (4 x b128 per lane).
        async_copy_b128(ldsA0, gA0 + k0);
        async_copy_b128(ldsA1, gA1 + k0);
        async_copy_b128(ldsB0, gB0 + k0);
        async_copy_b128(ldsB1, gB1 + k0);
        wait_async_all();
        __syncthreads();

        // Materialize all fragments, then an uninterrupted 8-WMMA burst.
        v16bf bfrag[2], afrag[4];
#pragma unroll
        for (int jn = 0; jn < 2; ++jn)
            bfrag[jn] = load_frag(Bs, wn * 32 + jn * 16 + r, 32, 0, hi);
#pragma unroll
        for (int im = 0; im < 4; ++im)
            afrag[im] = load_frag(As, wm * 64 + im * 16 + r, 32, 0, hi);
#pragma unroll
        for (int im = 0; im < 4; ++im)
#pragma unroll
            for (int jn = 0; jn < 2; ++jn)
                acc[im][jn] = __builtin_amdgcn_wmma_f32_16x16x32_bf16(
                    false, afrag[im], false, bfrag[jn], (short)0, acc[im][jn],
                    false, false);
        __syncthreads();
    }

    // Epilogue: lane holds C[hi*8 + j][r] of each 16x16 tile.
#pragma unroll
    for (int im = 0; im < 4; ++im) {
#pragma unroll
        for (int jn = 0; jn < 2; ++jn) {
            int ngl = n0 + wn * 32 + jn * 16 + r;
#pragma unroll
            for (int j = 0; j < 8; ++j) {
                int mgl = m0 + wm * 64 + im * 16 + hi * 8 + j;
                float val = acc[im][jn][j];
                if (mode == 0) {
                    C[(size_t)mgl * Ntotal + ngl] = val;
                } else if (ngl < Vc) {
                    int s = mgl >> 3, b = mgl & 7;
                    C[(size_t)(b * Sc + s) * Vc + ngl] = val + bout[ngl];
                }
            }
        }
    }
}

// ---------------------------------------------------------------------------
// One recurrence step:  g = xg[t] + h @ Uh + bias;  gate math; update h, c.
// 32 waves; wave w owns units [w*16, w*16+16) across ALL FOUR gates, so the
// full LSTM cell update happens in-register. M padded 8->16 (zero rows).
// UhT (2 MB bf16) stays L2-resident across all 256 steps.
// ---------------------------------------------------------------------------

__global__ __launch_bounds__(128) void lstm_step(
        const unsigned short* __restrict__ hIn, unsigned short* __restrict__ hOut,
        float* __restrict__ hF, float* __restrict__ cF,
        const unsigned short* __restrict__ UhT, const float* __restrict__ xg,
        const float* __restrict__ bias, unsigned short* __restrict__ hs, int t) {
    const int lane = threadIdx.x & 31;
    const int wave = blockIdx.x * 4 + (threadIdx.x >> 5);   // 0..31
    const int n0v  = wave * 16;
    const int r  = lane & 15, hi = lane >> 4;

    v8f acc[4];
#pragma unroll
    for (int g = 0; g < 4; ++g) acc[g] = (v8f){};

    for (int k0 = 0; k0 < 512; k0 += 32) {
        v16bf a = load_frag(hIn, r, 512, k0, hi);
        v16bf b[4];
#pragma unroll
        for (int g = 0; g < 4; ++g)
            b[g] = load_frag(UhT, g * 512 + n0v + r, 512, k0, hi);
#pragma unroll
        for (int g = 0; g < 4; ++g)
            acc[g] = __builtin_amdgcn_wmma_f32_16x16x32_bf16(
                false, a, false, b[g], (short)0, acc[g], false, false);
    }

    if (hi == 0) {                       // lanes 0..15 hold batch rows 0..7
        const int u = n0v + r;
        const float* xgt = xg + (size_t)(t * Bc) * Gc;
#pragma unroll
        for (int j = 0; j < Bc; ++j) {
            float gF = acc[0][j] + xgt[(size_t)j * Gc +            u] + bias[u];
            float gI = acc[1][j] + xgt[(size_t)j * Gc +  512 + u] + bias[512 + u];
            float gO = acc[2][j] + xgt[(size_t)j * Gc + 1024 + u] + bias[1024 + u];
            float gC = acc[3][j] + xgt[(size_t)j * Gc + 1536 + u] + bias[1536 + u];
            float c = sigm(gF) * cF[j * Hc + u] + sigm(gI) * tanhf(gC);
            float h = sigm(gO) * tanhf(c);
            cF[j * Hc + u] = c;
            hF[j * Hc + u] = h;
            unsigned short hb = f2bf(h);
            hOut[j * Hc + u] = hb;                       // ping-pong buffer
            hs[(size_t)(t * Bc + j) * Hc + u] = hb;      // archive for logits
        }
    }
}

__global__ void copy_tail(const float* __restrict__ hF, const float* __restrict__ cF,
                          float* __restrict__ out) {
    int idx = blockIdx.x * 256 + threadIdx.x;    // 0..4095
    size_t base = (size_t)Bc * Sc * Vc;
    out[base + idx]              = hF[idx];      // hT
    out[base + Bc * Hc + idx]    = cF[idx];      // cT
}

// ---------------------------------------------------------------------------

extern "C" void kernel_launch(void* const* d_in, const int* in_sizes, int n_in,
                              void* d_out, int out_size, void* d_ws, size_t ws_size,
                              hipStream_t stream) {
    (void)in_sizes; (void)n_in; (void)out_size; (void)ws_size;

    const int*   x    = (const int*)  d_in[0];
    const float* h0   = (const float*)d_in[1];
    const float* c0   = (const float*)d_in[2];
    const float* emb  = (const float*)d_in[3];
    const float* Wf   = (const float*)d_in[4];  const float* bWf = (const float*)d_in[5];
    const float* Uf   = (const float*)d_in[6];  const float* bUf = (const float*)d_in[7];
    const float* Wi   = (const float*)d_in[8];  const float* bWi = (const float*)d_in[9];
    const float* Ui   = (const float*)d_in[10]; const float* bUi = (const float*)d_in[11];
    const float* Wo   = (const float*)d_in[12]; const float* bWo = (const float*)d_in[13];
    const float* Uo   = (const float*)d_in[14]; const float* bUo = (const float*)d_in[15];
    const float* Wc   = (const float*)d_in[16]; const float* bWc = (const float*)d_in[17];
    const float* Uc   = (const float*)d_in[18]; const float* bUc = (const float*)d_in[19];
    const float* Wout = (const float*)d_in[20]; const float* bout = (const float*)d_in[21];
    float* out = (float*)d_out;

    // ---- workspace carve-up (256B aligned) ----
    char* ws = (char*)d_ws;
    size_t off = 0;
    auto carve = [&](size_t bytes) -> char* {
        char* p = ws + off;
        off = (off + bytes + 255) & ~(size_t)255;
        return p;
    };
    unsigned short* WxT   = (unsigned short*)carve((size_t)Gc * Ec * 2);     //  2 MB
    unsigned short* UhT   = (unsigned short*)carve((size_t)Gc * Hc * 2);     //  2 MB
    unsigned short* xe    = (unsigned short*)carve((size_t)Mrows * Hc * 2);  //  2 MB
    unsigned short* hs    = (unsigned short*)carve((size_t)Mrows * Hc * 2);  //  2 MB
    unsigned short* WoutT = (unsigned short*)carve((size_t)NPAD * Hc * 2);   // 51.5 MB
    float*          xg    = (float*)carve((size_t)Mrows * Gc * 4);           // 16 MB
    unsigned short* hb0   = (unsigned short*)carve((size_t)16 * Hc * 2);
    unsigned short* hb1   = (unsigned short*)carve((size_t)16 * Hc * 2);
    float*          hF    = (float*)carve((size_t)Bc * Hc * 4);
    float*          cF    = (float*)carve((size_t)Bc * Hc * 4);
    float*          bias  = (float*)carve((size_t)Gc * 4);

    // ---- setup ----
    build_WUT<<<Gc, 256, 0, stream>>>(Wf, Wi, Wo, Wc, Uf, Ui, Uo, Uc, WxT, UhT);
    build_WoutT<<<NPAD / 128, 128, 0, stream>>>(Wout, WoutT);
    init_bias<<<Gc / 256, 256, 0, stream>>>(bWf, bUf, bWi, bUi, bWo, bUo, bWc, bUc, bias);
    init_state<<<32, 256, 0, stream>>>(h0, c0, hb0, hb1, hF, cF);
    embed_gather<<<Mrows, 128, 0, stream>>>(x, emb, xe);

    // ---- input projection: xg = xe @ Wx   (M=2048, N=2048, K=512) ----
    gemm_bf16_wmma<<<dim3(Gc / 128, Mrows / 128), 256, 0, stream>>>(
        xe, WxT, xg, Gc, 0, nullptr);

    // ---- recurrence: 256 sequential steps, h ping-pongs hb0 <-> hb1 ----
    for (int t = 0; t < Sc; ++t) {
        const unsigned short* hin  = (t & 1) ? hb1 : hb0;
        unsigned short*       hout = (t & 1) ? hb0 : hb1;
        lstm_step<<<8, 128, 0, stream>>>(hin, hout, hF, cF, UhT, xg, bias, hs, t);
    }

    // ---- output projection: logits = hs @ Wout + bout  (N padded to 50304) ----
    gemm_bf16_wmma<<<dim3(NPAD / 128, Mrows / 128), 256, 0, stream>>>(
        hs, WoutT, out, NPAD, 1, bout);

    // ---- hT, cT tail ----
    copy_tail<<<16, 256, 0, stream>>>(hF, cF, out);
}
